// EdgeCompute_57801669870144
// MI455X (gfx1250) — compile-verified, matrix operand order checked by reading, not run on hardware
//
#include <hip/hip_runtime.h>
#include <hip/hip_bf16.h>
#include <math.h>

// CDNA5 / gfx1250, wave32. Edge-MLP: vals = sigmoid(relu(|x[src]-x[dst]| @ W1^T + b1) @ W2^T + b2)
// Layer-1 GEMM mapped to v_wmma_f32_16x16x32_f16: M=16 edges/wave, K=64 (2 steps), N=64 (4 tiles).

typedef __attribute__((ext_vector_type(16))) _Float16 v16h;
typedef __attribute__((ext_vector_type(8)))  float    v8f;

#define FEAT 64
#define HID  64

// |v| via explicit sign-bit clear in inline asm. This is deliberately opaque:
// every C-level spelling of fabs (fabsf, select, fmax(x,-x), bit-and in C)
// gets canonicalized to llvm.fabs, which the gfx1250 backend miscompiles when
// SLP vectorizes it to v2f32 (invalid v2f32<->i32 bitcast in ISel).
__device__ __forceinline__ float abs_opaque(float v) {
    float r;
    asm("v_and_b32 %0, 0x7fffffff, %1" : "=v"(r) : "v"(v));
    return r;
}

__global__ __launch_bounds__(256) void edge_mlp_wmma(
    const float* __restrict__ x,      // [N_NODES, 64]
    const int*   __restrict__ ei,     // [2, E]
    const float* __restrict__ W1,     // [64, 64]
    const float* __restrict__ b1,     // [64]
    const float* __restrict__ W2,     // [64]  (W2 is [1,64])
    const float* __restrict__ b2,     // [1]
    float*       __restrict__ out,    // [E]
    int nEdges)
{
    // ---- Stage W1 into LDS as f16 (once per block) ----
    __shared__ _Float16 sW1[HID * FEAT];   // 8 KB
    for (int i = threadIdx.x; i < HID * FEAT; i += blockDim.x)
        sW1[i] = (_Float16)W1[i];
    __syncthreads();

    const int lane = threadIdx.x & 31;
    const int m    = lane & 15;     // row (edge-in-tile) for A; col n_local for B
    const int hi   = lane >> 4;     // which K half-group this lane carries

    const int wavesPerBlock = blockDim.x >> 5;
    const int waveId = blockIdx.x * wavesPerBlock + (threadIdx.x >> 5);
    const int nWaves = gridDim.x * wavesPerBlock;

    // ---- Loop-invariant B fragments: B[t][s] covers N-tile t (n = t*16 + m), K-step s ----
    v16h B[4][2];
    #pragma unroll
    for (int t = 0; t < 4; ++t) {
        const _Float16* wrow = &sW1[(t * 16 + m) * FEAT];
        #pragma unroll
        for (int s = 0; s < 2; ++s) {
            const int k0 = s * 32 + hi * 8;
            const int k1 = s * 32 + 16 + hi * 8;
            #pragma unroll
            for (int j = 0; j < 8; ++j) {
                B[t][s][j]     = wrow[k0 + j];
                B[t][s][8 + j] = wrow[k1 + j];
            }
        }
    }

    // per-lane layer-2 constants (n = t*16 + m for lane's column)
    float b1v[4], w2v[4];
    #pragma unroll
    for (int t = 0; t < 4; ++t) {
        b1v[t] = b1[t * 16 + m];
        w2v[t] = W2[t * 16 + m];
    }
    const float bias2 = b2[0];

    const int nTiles = (nEdges + 15) >> 4;
    for (int tile = waveId; tile < nTiles; tile += nWaves) {
        // Each lane handles edge (tile*16 + m); both halves (hi=0/1) load same edge ids.
        const int e  = tile * 16 + m;
        const int ec = (e < nEdges) ? e : 0;              // lane-select, no EXEC change
        const int src = ei[ec];
        const int dst = ei[nEdges + ec];
        const float* xs = x + (size_t)src * FEAT;
        const float* xd = x + (size_t)dst * FEAT;

        v8f C[4];
        #pragma unroll
        for (int t = 0; t < 4; ++t)
            C[t] = (v8f){0.f, 0.f, 0.f, 0.f, 0.f, 0.f, 0.f, 0.f};

        #pragma unroll
        for (int s = 0; s < 2; ++s) {
            // A fragment: |x[src][k] - x[dst][k]| -> f16, ISA 16-bit A-layout.
            const int k0 = s * 32 + hi * 8;
            const int k1 = s * 32 + 16 + hi * 8;
            v16h A;
            #pragma unroll
            for (int j = 0; j < 8; ++j) {
                float d0 = xs[k0 + j] - xd[k0 + j];
                float d1 = xs[k1 + j] - xd[k1 + j];
                A[j]     = (_Float16)abs_opaque(d0);
                A[8 + j] = (_Float16)abs_opaque(d1);
            }
            #pragma unroll
            for (int t = 0; t < 4; ++t) {
                // D = A*B + C ; emits v_wmma_f32_16x16x32_f16
                C[t] = __builtin_amdgcn_wmma_f32_16x16x32_f16(
                    /*neg_a=*/false, A, /*neg_b=*/false, B[t][s],
                    /*c_mod=*/(short)0, C[t], /*reuse_a=*/false, /*reuse_b=*/false);
            }
        }

        // ---- Layer 2: acc[r] = sum_n relu(h[M][n] + b1[n]) * W2[n] ----
        // C layout: lane group (0-15 / 16-31), VGPR r -> M = hi*8 + r, N = t*16 + m.
        float acc[8];
        #pragma unroll
        for (int r = 0; r < 8; ++r) acc[r] = 0.f;
        #pragma unroll
        for (int t = 0; t < 4; ++t) {
            const float w2 = w2v[t];
            const float bb = b1v[t];
            #pragma unroll
            for (int r = 0; r < 8; ++r) {
                float hv = C[t][r] + bb;
                hv = hv > 0.f ? hv : 0.f;
                acc[r] += hv * w2;
            }
        }
        // Butterfly-reduce over the 16 lanes of each half (xor masks 1,2,4,8 stay in-group)
        #pragma unroll
        for (int r = 0; r < 8; ++r) {
            float v = acc[r];
            #pragma unroll
            for (int d = 1; d < 16; d <<= 1)
                v += __shfl_xor(v, d, 32);
            acc[r] = v;
        }

        // Lanes with m<8 each own one output row: M = hi*8 + m
        if (m < 8) {
            const int M  = hi * 8 + m;
            const int eo = tile * 16 + M;
            if (eo < nEdges) {
                const float sgm = acc[m] + bias2;
                out[eo] = 1.0f / (1.0f + __expf(-sgm));
            }
        }
    }
}

extern "C" void kernel_launch(void* const* d_in, const int* in_sizes, int n_in,
                              void* d_out, int out_size, void* d_ws, size_t ws_size,
                              hipStream_t stream) {
    const float* x  = (const float*)d_in[0];
    const int*   ei = (const int*)  d_in[1];
    const float* W1 = (const float*)d_in[2];
    const float* b1 = (const float*)d_in[3];
    const float* W2 = (const float*)d_in[4];
    const float* b2 = (const float*)d_in[5];
    float* out = (float*)d_out;

    const int nEdges = in_sizes[1] / 2;         // edge_index is [2, E]
    const int threads = 256;                    // 8 wave32 per block
    const int wavesPerBlock = threads / 32;
    const int nTiles = (nEdges + 15) / 16;
    int blocks = (nTiles + wavesPerBlock - 1) / wavesPerBlock;
    if (blocks > 1024) blocks = 1024;           // persistent grid, ~6 tiles/wave
    if (blocks < 1) blocks = 1;

    edge_mlp_wmma<<<blocks, threads, 0, stream>>>(x, ei, W1, b1, W2, b2, out, nEdges);
}
